// PredictionHead_10471130268021
// MI455X (gfx1250) — compile-verified, matrix operand order checked by reading, not run on hardware
//
#include <hip/hip_runtime.h>

// ---------------- types ----------------
typedef _Float16 half8 __attribute__((ext_vector_type(8)));
typedef _Float16 v16h  __attribute__((ext_vector_type(16)));
typedef float    v8f   __attribute__((ext_vector_type(8)));
typedef unsigned int uint4v __attribute__((ext_vector_type(4)));
typedef unsigned int uint8v __attribute__((ext_vector_type(8)));

#define N_CAND 4096
#define DIM    768
#define HID    1024
#define EDIM   7
#define RDIM   6
#define TOPC   256   // c = max(int(2560*0.1),1) = 256 (static in reference)

#define XPAD 776     // 768 + 8 f16 pad -> 1552B row stride (4 dwords mod 64 banks)
#define SPAD 40      // 32 + 8 f16 pad  -> 80B row stride (20 dwords mod 64 banks)

// ---------------- stage 1: mention FFNN + softmax[0] ----------------
__global__ __launch_bounds__(256) void mention_kernel(
    const float* __restrict__ emb, const float* __restrict__ Wm1,
    const float* __restrict__ bm1, const float* __restrict__ Wm2,
    const float* __restrict__ bm2, float* __restrict__ notE)
{
  __shared__ float xrow[DIM];
  __shared__ float red[256 * EDIM];
  const int row = blockIdx.x, t = threadIdx.x;
  for (int k = t; k < DIM; k += 256) xrow[k] = emb[row * DIM + k];
  __syncthreads();
  float part[EDIM];
  #pragma unroll
  for (int e = 0; e < EDIM; ++e) part[e] = 0.f;
  for (int u = 0; u < 4; ++u) {
    const int h = u * 256 + t;                  // coalesced over t
    float acc = bm1[h];
    for (int k = 0; k < DIM; ++k) acc = fmaf(xrow[k], Wm1[k * HID + h], acc);
    acc = fmaxf(acc, 0.f);
    #pragma unroll
    for (int e = 0; e < EDIM; ++e) part[e] = fmaf(acc, Wm2[h * EDIM + e], part[e]);
  }
  #pragma unroll
  for (int e = 0; e < EDIM; ++e) red[t * EDIM + e] = part[e];
  __syncthreads();
  for (int s = 128; s > 0; s >>= 1) {
    if (t < s)
      for (int e = 0; e < EDIM; ++e) red[t * EDIM + e] += red[(t + s) * EDIM + e];
    __syncthreads();
  }
  if (t == 0) {
    float sc[EDIM], mx = -1e30f;
    #pragma unroll
    for (int e = 0; e < EDIM; ++e) { sc[e] = red[e] + bm2[e]; mx = fmaxf(mx, sc[e]); }
    float sum = 0.f;
    #pragma unroll
    for (int e = 0; e < EDIM; ++e) sum += __expf(sc[e] - mx);
    notE[row] = __expf(sc[0] - mx) / sum;
  }
}

// ---------------- stage 2: deterministic top-256 by rank ----------------
__global__ __launch_bounds__(256) void rank_kernel(const float* __restrict__ notE,
                                                   int* __restrict__ flags)
{
  const int i = blockIdx.x * 256 + threadIdx.x;
  const float si = notE[i];
  int rank = 0;
  for (int j = 0; j < N_CAND; ++j) {
    const float sj = notE[j];
    rank += (sj < si) || (sj == si && j < i);   // total order -> exactly TOPC winners
  }
  flags[i] = (rank < TOPC) ? 1 : 0;
}

__global__ void compact_kernel(const int* __restrict__ flags, int* __restrict__ idx)
{
  if (threadIdx.x == 0 && blockIdx.x == 0) {
    int pos = 0;
    for (int i = 0; i < N_CAND && pos < TOPC; ++i)
      if (flags[i]) idx[pos++] = i;             // ascending index order == jnp.sort(idx)
  }
}

// ---------------- stage 3: gather top embeddings (f32 + f16) ----------------
__global__ __launch_bounds__(256) void gather_kernel(
    const float* __restrict__ emb, const int* __restrict__ idx,
    float* __restrict__ topf, _Float16* __restrict__ toph)
{
  const int r = blockIdx.x, g = idx[r];
  for (int k = threadIdx.x; k < DIM; k += 256) {
    const float v = emb[(size_t)g * DIM + k];
    topf[r * DIM + k] = v;
    toph[r * DIM + k] = (_Float16)v;
  }
}

// ---------------- stage 4: transpose+convert product block of Wr1 ----------------
// wr1h[h*DIM + k] = f16( Wr1[(2*DIM + k)*HID + h] )
__global__ __launch_bounds__(256) void wr1h_kernel(const float* __restrict__ Wr1,
                                                   _Float16* __restrict__ wr1h)
{
  const int e = blockIdx.x * 256 + threadIdx.x;     // e < HID*DIM
  const int h = e / DIM, k = e - h * DIM;
  wr1h[e] = (_Float16)Wr1[(size_t)(2 * DIM + k) * HID + h];
}

// ---------------- stage 5: Asrc' = x@Wr1_src + br1 ; Atgt = x@Wr1_tgt ----------------
__global__ __launch_bounds__(256) void head_kernel(
    const float* __restrict__ topf, const float* __restrict__ Wr1,
    const float* __restrict__ br1, float* __restrict__ Asrc, float* __restrict__ Atgt)
{
  __shared__ float xrow[DIM];
  const int j = blockIdx.x, t = threadIdx.x;
  for (int k = t; k < DIM; k += 256) xrow[k] = topf[j * DIM + k];
  __syncthreads();
  for (int u = 0; u < 4; ++u) {
    const int h = u * 256 + t;
    float a1 = br1[h], a2 = 0.f;
    for (int k = 0; k < DIM; ++k) {
      a1 = fmaf(xrow[k], Wr1[(size_t)k * HID + h], a1);
      a2 = fmaf(xrow[k], Wr1[(size_t)(DIM + k) * HID + h], a2);
    }
    Asrc[j * HID + h] = a1;
    Atgt[j * HID + h] = a2;
  }
}

// ---------------- TDM: 2D tile load (32 k x 64 h, f16) global -> LDS ----------------
// Tensor: wr1h viewed as [HID lines][DIM elems], data_size=2B, line stride DIM.
// Tile: tile_dim0=32 (64B = 16 dwords per line), tile_dim1=64 lines.
// LDS padding: pad_interval=16 dwords (code 3), pad_amount=4 dwords (code 3)
//   -> 80B row stride in LDS == SPAD(40) f16, matching slab layout exactly.
__device__ __forceinline__ void tdm_slab_load(const _Float16* gsrc, unsigned lds_off)
{
  const unsigned long long ga = (unsigned long long)(uintptr_t)gsrc;
  uint4v g0;
  g0[0] = 1u;                                           // count=1 (valid user D#)
  g0[1] = lds_off;                                      // lds_addr (bytes)
  g0[2] = (unsigned)(ga & 0xffffffffu);                 // global_addr[31:0]
  g0[3] = (unsigned)((ga >> 32) & 0x01ffffffu)          // global_addr[56:32]
        | (2u << 30);                                   // type=2 ("image")
  uint8v g1;
  g1[0] = (1u << 16)    // data_size = 2 bytes
        | (1u << 20)    // pad_enable
        | (3u << 22)    // pad_interval: every 16 dwords
        | (3u << 25);   // pad_amount: 4 dwords
  g1[1] = ((unsigned)DIM & 0xffffu) << 16;              // tensor_dim0[15:0]
  g1[2] = ((unsigned)HID & 0xffffu) << 16;              // dim0[31:16]=0 | tensor_dim1[15:0]
  g1[3] = (32u << 16);                                  // dim1[31:16]=0 | tile_dim0=32
  g1[4] = 64u;                                          // tile_dim1=64 | tile_dim2=0
  g1[5] = (unsigned)DIM;                                // tensor_dim0_stride[31:0]
  g1[6] = 0u;                                           // stride0 hi | stride1 lo
  g1[7] = 0u;
  asm volatile("tensor_load_to_lds %0, %1" :: "s"(g0), "s"(g1) : "memory");
}

// ---------------- stage 6: fused pair FFNN (WMMA + TDM double-buffer) ----------------
// grid (16,16). 512 threads = 16 waves; wave w -> i = i0+w, WMMA M-dim = 16 j's.
// hid[i,j,h] = Asrc'[j,h] + Atgt[i,h] + sum_k (xi[k]*xj[k]) * Wr1_prod[k,h]
// out[i,j,r] = sum_h relu(hid) * Wr2[h,r] + br2[r]
__global__ __launch_bounds__(512) void pair_kernel(
    const _Float16* __restrict__ toph, const _Float16* __restrict__ wr1h,
    const float* __restrict__ Asrc,   const float* __restrict__ Atgt,
    const float* __restrict__ Wr2,    const float* __restrict__ br2,
    float* __restrict__ out)
{
  __shared__ _Float16 xi[16 * XPAD];
  __shared__ _Float16 xj[16 * XPAD];
  __shared__ _Float16 slab[2][64 * SPAD];   // double-buffered TDM destination

  const int tid  = threadIdx.x;
  const int w    = tid >> 5;            // wave id 0..15
  const int lane = tid & 31;
  const int half = (lane >> 4) & 1;
  const int n    = lane & 15;
  const int i0 = blockIdx.y * 16, j0 = blockIdx.x * 16;

  // stage xi/xj tiles (f16) into LDS
  for (int e = tid; e < 16 * DIM; e += 512) {
    const int r = e / DIM, k = e - r * DIM;
    xi[r * XPAD + k] = toph[(i0 + r) * DIM + k];
    xj[r * XPAD + k] = toph[(j0 + r) * DIM + k];
  }

  const unsigned slab_off0 = (unsigned)(uintptr_t)&slab[0][0];
  const unsigned slab_off1 = (unsigned)(uintptr_t)&slab[1][0];

  float op[8][RDIM];
  #pragma unroll
  for (int r = 0; r < 8; ++r)
    #pragma unroll
    for (int rr = 0; rr < RDIM; ++rr) op[r][rr] = 0.f;

  const int irow = i0 + w;

  for (int chunk = 0; chunk < 16; ++chunk) {
    v8f acc[4];
    #pragma unroll
    for (int t4 = 0; t4 < 4; ++t4)
      #pragma unroll
      for (int r = 0; r < 8; ++r) acc[t4][r] = 0.f;

    const _Float16* wbase = &wr1h[(size_t)(chunk * 64) * DIM];

    __syncthreads();                          // both buffers free; xi/xj staged
    if (w == 0) tdm_slab_load(wbase, slab_off0);   // prefetch kb=0 -> buf0

    for (int kb = 0; kb < 24; ++kb) {
      const int cur = kb & 1;
      if (kb + 1 < 24) {
        __syncthreads();                      // everyone done reading buf[!cur]
        if (w == 0)
          tdm_slab_load(wbase + (kb + 1) * 32, cur ? slab_off0 : slab_off1);
      }
      if (w == 0) {
        if (kb < 23) __builtin_amdgcn_s_wait_tensorcnt(1);   // buf[cur] landed
        else         __builtin_amdgcn_s_wait_tensorcnt(0);
      }
      __syncthreads();                        // slab kb visible to all waves

      // A fragment: 16 j-rows x 32 k of (xi .* xj), row M = lane%16
      const _Float16* xip = &xi[w * XPAD + kb * 32 + 8 * half];
      const _Float16* xjp = &xj[n * XPAD + kb * 32 + 8 * half];
      const half8 a0 = *(const half8*)xip        * *(const half8*)xjp;
      const half8 a1 = *(const half8*)(xip + 16) * *(const half8*)(xjp + 16);
      union { v16h v; half8 h[2]; } af;
      af.h[0] = a0; af.h[1] = a1;

      const _Float16* sb = &slab[cur][0];
      #pragma unroll
      for (int t4 = 0; t4 < 4; ++t4) {
        const _Float16* bp = &sb[(t4 * 16 + n) * SPAD + 8 * half];
        union { v16h v; half8 h[2]; } bf;
        bf.h[0] = *(const half8*)bp;
        bf.h[1] = *(const half8*)(bp + 16);
        acc[t4] = __builtin_amdgcn_wmma_f32_16x16x32_f16(
            false, af.v, false, bf.v, (short)0, acc[t4], false, false);
      }
    }

    // epilogue: bias/linear terms + relu + fold layer 2 into per-lane partials
    #pragma unroll
    for (int t4 = 0; t4 < 4; ++t4) {
      const int h = chunk * 64 + t4 * 16 + n;
      const float atv = Atgt[(size_t)irow * HID + h];
      float w2[RDIM];
      #pragma unroll
      for (int rr = 0; rr < RDIM; ++rr) w2[rr] = Wr2[h * RDIM + rr];
      #pragma unroll
      for (int r = 0; r < 8; ++r) {
        const int m = r + 8 * half;                       // j-row slot of this lane
        float val = acc[t4][r] + Asrc[(size_t)(j0 + m) * HID + h] + atv;
        val = fmaxf(val, 0.f);
        #pragma unroll
        for (int rr = 0; rr < RDIM; ++rr) op[r][rr] = fmaf(val, w2[rr], op[r][rr]);
      }
    }
  }

  // cross-lane reduction over hidden-column lanes (n = 0..15), stays within half
  #pragma unroll
  for (int mask = 1; mask <= 8; mask <<= 1)
    #pragma unroll
    for (int r = 0; r < 8; ++r)
      #pragma unroll
      for (int rr = 0; rr < RDIM; ++rr)
        op[r][rr] += __shfl_xor(op[r][rr], mask, 32);

  if (n == 0) {
    #pragma unroll
    for (int r = 0; r < 8; ++r) {
      const int jg = j0 + r + 8 * half;
      float* dst = &out[((size_t)irow * TOPC + jg) * RDIM];
      #pragma unroll
      for (int rr = 0; rr < RDIM; ++rr) dst[rr] = op[r][rr] + br2[rr];
    }
  }
}

// ---------------- launch ----------------
extern "C" void kernel_launch(void* const* d_in, const int* in_sizes, int n_in,
                              void* d_out, int out_size, void* d_ws, size_t ws_size,
                              hipStream_t stream)
{
  const float* emb = (const float*)d_in[0];
  const float* Wm1 = (const float*)d_in[1];
  const float* bm1 = (const float*)d_in[2];
  const float* Wm2 = (const float*)d_in[3];
  const float* bm2 = (const float*)d_in[4];
  const float* Wr1 = (const float*)d_in[5];
  const float* br1 = (const float*)d_in[6];
  const float* Wr2 = (const float*)d_in[7];
  const float* br2 = (const float*)d_in[8];
  float* out = (float*)d_out;

  char* p = (char*)d_ws;
  auto carve = [&](size_t bytes) { char* r = p; p += (bytes + 255) & ~(size_t)255; return r; };
  float*    notE  = (float*)   carve((size_t)N_CAND * 4);
  int*      flags = (int*)     carve((size_t)N_CAND * 4);
  int*      idx   = (int*)     carve((size_t)TOPC * 4);
  float*    topf  = (float*)   carve((size_t)TOPC * DIM * 4);
  _Float16* toph  = (_Float16*)carve((size_t)TOPC * DIM * 2);
  _Float16* wr1h  = (_Float16*)carve((size_t)HID * DIM * 2);
  float*    Asrc  = (float*)   carve((size_t)TOPC * HID * 4);
  float*    Atgt  = (float*)   carve((size_t)TOPC * HID * 4);

  mention_kernel<<<N_CAND, 256, 0, stream>>>(emb, Wm1, bm1, Wm2, bm2, notE);
  rank_kernel<<<N_CAND / 256, 256, 0, stream>>>(notE, flags);
  compact_kernel<<<1, 32, 0, stream>>>(flags, idx);
  gather_kernel<<<TOPC, 256, 0, stream>>>(emb, idx, topf, toph);
  wr1h_kernel<<<(HID * DIM) / 256, 256, 0, stream>>>(Wr1, wr1h);
  head_kernel<<<TOPC, 256, 0, stream>>>(topf, Wr1, br1, Asrc, Atgt);
  pair_kernel<<<dim3(TOPC / 16, TOPC / 16), 512, 0, stream>>>(
      toph, wr1h, Asrc, Atgt, Wr2, br2, out);
}
